// AttentionGuidedGate_8108898255610
// MI455X (gfx1250) — compile-verified, matrix operand order checked by reading, not run on hardware
//
#include <hip/hip_runtime.h>
#include <hip/hip_bf16.h>
#include <math.h>

typedef unsigned short u16;
typedef __attribute__((ext_vector_type(16))) __bf16 v16bf;
typedef __attribute__((ext_vector_type(8)))  __bf16 v8bf;
typedef __attribute__((ext_vector_type(8)))  float  v8f;

#define T_TOK 2048
#define DIM_  1024
#define AH_   2048
#define HID_  1024
#define NEXP  8
#define NHEAD 16
#define HDIM  128
#define QKN   2048   /* NHEAD*HDIM */

// ---------- helpers ----------
__device__ __forceinline__ u16 f2bf(float f) {
  union { float f; unsigned u; } v; v.f = f;
  unsigned r = v.u + 0x7FFFu + ((v.u >> 16) & 1u);   // round-to-nearest-even
  return (u16)(r >> 16);
}
__device__ __forceinline__ float bf2f(u16 h) {
  union { unsigned u; float f; } v; v.u = ((unsigned)h) << 16;
  return v.f;
}
// Build a 16x32 bf16 A/B fragment slice for this lane.
// Lane half hi=lane/16 holds K = {8*hi..8*hi+7} and {16+8*hi..23+8*hi}
__device__ __forceinline__ v16bf load_frag(const u16* p, int hi) {
  v8bf lo = *reinterpret_cast<const v8bf*>(p + 8 * hi);
  v8bf h8 = *reinterpret_cast<const v8bf*>(p + 16 + 8 * hi);
  v16bf r;
#pragma unroll
  for (int i = 0; i < 8; ++i) { r[i] = lo[i]; r[i + 8] = h8[i]; }
  return r;
}
__device__ __forceinline__ float gelu_exact(float v) {
  return 0.5f * v * (1.0f + erff(v * 0.70710678118654752f));
}

// ---------- tiny utility kernels ----------
__global__ void zero_f32_kernel(float* p, int n) {
  int i = blockIdx.x * 256 + threadIdx.x;
  if (i < n) p[i] = 0.0f;
}
__global__ void cvt_f32_bf16_kernel(const float* __restrict__ in, u16* __restrict__ out, int n) {
  int i = blockIdx.x * 256 + threadIdx.x;
  if (i < n) out[i] = f2bf(in[i]);
}
// in: [K,N] f32 row-major  ->  out: [N,K] bf16 row-major (N-major for WMMA B frags)
__global__ void transpose_f32_bf16_kernel(const float* __restrict__ in, u16* __restrict__ out,
                                          int K, int N) {
  int i = blockIdx.x * 256 + threadIdx.x;
  if (i >= K * N) return;
  int k = i / N, n = i - k * N;
  out[(size_t)n * K + k] = f2bf(in[i]);
}
// per-row rmsnorm (row length = cols), scaled by w, output bf16
__global__ __launch_bounds__(256) void rmsnorm_rows_kernel(const float* __restrict__ in,
                                                           const float* __restrict__ w,
                                                           u16* __restrict__ out, int cols) {
  __shared__ float red[256];
  const int row = blockIdx.x, tid = threadIdx.x;
  const float* p = in + (size_t)row * cols;
  float ss = 0.f;
  for (int c = tid; c < cols; c += 256) { float v = p[c]; ss += v * v; }
  red[tid] = ss; __syncthreads();
  for (int s = 128; s; s >>= 1) { if (tid < s) red[tid] += red[tid + s]; __syncthreads(); }
  float scale = rsqrtf(red[0] / (float)cols + 1e-6f);
  u16* o = out + (size_t)row * cols;
  for (int c = tid; c < cols; c += 256) o[c] = f2bf(p[c] * scale * w[c]);
}
// per-(token,head) rmsnorm on 128-d head vectors, f32 in -> bf16 out
__global__ __launch_bounds__(32) void head_rmsnorm_kernel(const float* __restrict__ in,
                                                          const float* __restrict__ w,
                                                          u16* __restrict__ out) {
  const int b = blockIdx.x;            // t*NHEAD + h
  const int t = b >> 4, h = b & 15;
  const int lane = threadIdx.x & 31;
  const float* p = in + (size_t)t * QKN + h * HDIM;
  float4 v = *reinterpret_cast<const float4*>(p + lane * 4);
  float ss = v.x * v.x + v.y * v.y + v.z * v.z + v.w * v.w;
#pragma unroll
  for (int off = 16; off; off >>= 1) ss += __shfl_xor(ss, off);
  float scale = rsqrtf(ss * (1.0f / HDIM) + 1e-6f);
  int d = lane * 4;
  u16* o = out + (size_t)t * QKN + h * HDIM + d;
  o[0] = f2bf(v.x * scale * w[d + 0]);
  o[1] = f2bf(v.y * scale * w[d + 1]);
  o[2] = f2bf(v.z * scale * w[d + 2]);
  o[3] = f2bf(v.w * scale * w[d + 3]);
}

// ---------- bf16 WMMA GEMM:  C[M,N] = A[M,K] * BT[N,K]^T (+bias) ----------
// block = 256 threads = 8 waves; each wave: 16 rows x 64 cols; block: 128 x 64
// K-loop is software-pipelined: fragments for step k+32 are issued before the
// 4 WMMAs of step k, so global_load_b128 latency overlaps the matrix pipe.
// mode 0: f32 out (+bias);  mode 1: exact-GELU -> bf16 out (+bias)
__global__ __launch_bounds__(256) void gemm_bf16_kernel(const u16* __restrict__ A,
                                                        const u16* __restrict__ BT,
                                                        const float* __restrict__ bias,
                                                        float* __restrict__ Cf,
                                                        u16* __restrict__ Cb,
                                                        int N, int Kd, int mode) {
  const int lane = threadIdx.x & 31, wave = threadIdx.x >> 5;
  const int hi = lane >> 4, l16 = lane & 15;
  const int m0 = blockIdx.y * 128 + wave * 16;
  const int n0 = blockIdx.x * 64;
  const u16* Arow = A + (size_t)(m0 + l16) * Kd;
  const u16* Brow0 = BT + (size_t)(n0 + 0 * 16 + l16) * Kd;
  const u16* Brow1 = BT + (size_t)(n0 + 1 * 16 + l16) * Kd;
  const u16* Brow2 = BT + (size_t)(n0 + 2 * 16 + l16) * Kd;
  const u16* Brow3 = BT + (size_t)(n0 + 3 * 16 + l16) * Kd;

  v8f acc[4] = {v8f{}, v8f{}, v8f{}, v8f{}};

  // prologue: fragments for k = 0
  v16bf a_cur  = load_frag(Arow, hi);
  v16bf b_cur0 = load_frag(Brow0, hi);
  v16bf b_cur1 = load_frag(Brow1, hi);
  v16bf b_cur2 = load_frag(Brow2, hi);
  v16bf b_cur3 = load_frag(Brow3, hi);

  for (int kk = 0; kk + 32 < Kd; kk += 32) {
    const int kn = kk + 32;
    // issue next step's loads first (in-flight across the WMMA burst)
    v16bf a_nxt  = load_frag(Arow + kn, hi);
    v16bf b_nxt0 = load_frag(Brow0 + kn, hi);
    v16bf b_nxt1 = load_frag(Brow1 + kn, hi);
    v16bf b_nxt2 = load_frag(Brow2 + kn, hi);
    v16bf b_nxt3 = load_frag(Brow3 + kn, hi);
    // matrix pipe on current fragments
    acc[0] = __builtin_amdgcn_wmma_f32_16x16x32_bf16(false, a_cur, false, b_cur0, (short)0, acc[0], false, false);
    acc[1] = __builtin_amdgcn_wmma_f32_16x16x32_bf16(false, a_cur, false, b_cur1, (short)0, acc[1], false, false);
    acc[2] = __builtin_amdgcn_wmma_f32_16x16x32_bf16(false, a_cur, false, b_cur2, (short)0, acc[2], false, false);
    acc[3] = __builtin_amdgcn_wmma_f32_16x16x32_bf16(false, a_cur, false, b_cur3, (short)0, acc[3], false, false);
    a_cur = a_nxt; b_cur0 = b_nxt0; b_cur1 = b_nxt1; b_cur2 = b_nxt2; b_cur3 = b_nxt3;
  }
  // epilogue WMMA for the final k-step
  acc[0] = __builtin_amdgcn_wmma_f32_16x16x32_bf16(false, a_cur, false, b_cur0, (short)0, acc[0], false, false);
  acc[1] = __builtin_amdgcn_wmma_f32_16x16x32_bf16(false, a_cur, false, b_cur1, (short)0, acc[1], false, false);
  acc[2] = __builtin_amdgcn_wmma_f32_16x16x32_bf16(false, a_cur, false, b_cur2, (short)0, acc[2], false, false);
  acc[3] = __builtin_amdgcn_wmma_f32_16x16x32_bf16(false, a_cur, false, b_cur3, (short)0, acc[3], false, false);

#pragma unroll
  for (int j = 0; j < 4; ++j) {
    int n = n0 + j * 16 + l16;
    float bv = bias ? bias[n] : 0.0f;
#pragma unroll
    for (int r = 0; r < 8; ++r) {
      int m = m0 + r + 8 * hi;                  // C layout: VGPR r, lane-half hi
      float v = acc[j][r] + bv;
      if (mode == 1) Cb[(size_t)m * N + n] = f2bf(gelu_exact(v));
      else           Cf[(size_t)m * N + n] = v;
    }
  }
}

// ---------- attention scores + softmax + importance (flash-style, LDS-resident) ----------
// grid: (qtile=128, head=16); block 256; dynamic LDS = 16*2048*4 = 128 KB (CDNA5 WGP has 320 KB)
// Key-tile sweep is double-buffered the same way as the GEMM K-loop.
__global__ __launch_bounds__(256) void scores_imp_kernel(const u16* __restrict__ Qb,
                                                         const u16* __restrict__ Kb,
                                                         float* __restrict__ imp_h) {
  extern __shared__ float srow[];               // [16][2048] score rows
  __shared__ float mrow[16], zrow[16];
  const int lane = threadIdx.x & 31, wave = threadIdx.x >> 5;
  const int hi = lane >> 4, l16 = lane & 15;
  const int qt = blockIdx.x, head = blockIdx.y;

  // A fragments: Q tile [16 x 128], 4 k-slices of 32, resident in VGPRs
  const u16* Qrow = Qb + (size_t)(qt * 16 + l16) * QKN + head * HDIM;
  v16bf a[4];
#pragma unroll
  for (int ks = 0; ks < 4; ++ks) a[ks] = load_frag(Qrow + ks * 32, hi);

  // each wave sweeps key tiles kt = wave, wave+8, ... (16 tiles per wave)
  const size_t krow_stride = (size_t)16 * QKN * 8;   // advance 8 key tiles
  const u16* Krow = Kb + (size_t)(wave * 16 + l16) * QKN + head * HDIM;
  v16bf b_cur[4];
#pragma unroll
  for (int ks = 0; ks < 4; ++ks) b_cur[ks] = load_frag(Krow + ks * 32, hi);

  for (int kt = wave; kt < T_TOK / 16; kt += 8) {
    v16bf b_nxt[4];
    if (kt + 8 < T_TOK / 16) {
      const u16* Knext = Krow + krow_stride;
#pragma unroll
      for (int ks = 0; ks < 4; ++ks) b_nxt[ks] = load_frag(Knext + ks * 32, hi);
    }
    v8f acc = v8f{};
#pragma unroll
    for (int ks = 0; ks < 4; ++ks)
      acc = __builtin_amdgcn_wmma_f32_16x16x32_bf16(false, a[ks], false, b_cur[ks],
                                                    (short)0, acc, false, false);
#pragma unroll
    for (int r = 0; r < 8; ++r)
      srow[(r + 8 * hi) * T_TOK + kt * 16 + l16] = acc[r] * 0.08838834764831845f; // 1/sqrt(128)
    Krow += krow_stride;
#pragma unroll
    for (int ks = 0; ks < 4; ++ks) b_cur[ks] = b_nxt[ks];
  }
  __syncthreads();

  // online softmax stats: wave w owns query rows 2w, 2w+1
#pragma unroll
  for (int rr = 0; rr < 2; ++rr) {
    int q = wave * 2 + rr;
    float m = -3.0e38f, sum = 0.f;
    for (int c = lane; c < T_TOK; c += 32) {
      float v = srow[q * T_TOK + c];
      float mn = fmaxf(m, v);
      sum = sum * __expf(m - mn) + __expf(v - mn);
      m = mn;
    }
#pragma unroll
    for (int off = 16; off; off >>= 1) {
      float m2 = __shfl_xor(m, off);
      float s2 = __shfl_xor(sum, off);
      float mn = fmaxf(m, m2);
      sum = sum * __expf(m - mn) + s2 * __expf(m2 - mn);
      m = mn;
    }
    if (lane == 0) { mrow[q] = m; zrow[q] = 1.0f / sum; }
  }
  __syncthreads();

  // importance: sum probs over the 16 query rows, accumulate per head
  for (int c = threadIdx.x; c < T_TOK; c += 256) {
    float local = 0.f;
#pragma unroll
    for (int q = 0; q < 16; ++q)
      local += __expf(srow[q * T_TOK + c] - mrow[q]) * zrow[q];
    atomicAdd(&imp_h[(size_t)head * T_TOK + c], local);
  }
}

__global__ void finalize_imp_kernel(const float* __restrict__ imp_h, float* __restrict__ out) {
  int t = blockIdx.x * 256 + threadIdx.x;
  if (t >= T_TOK) return;
  float s = 0.f;
#pragma unroll
  for (int h = 0; h < NHEAD; ++h) s += imp_h[(size_t)h * T_TOK + t];
  out[2 * T_TOK + t] = s * (1.0f / NHEAD);
}

// ---------- router tail: logits (E=8), softmax, group-limited top-1 ----------
__global__ __launch_bounds__(32) void router_kernel(const u16* __restrict__ h2,
                                                    const float* __restrict__ w3,
                                                    const float* __restrict__ b3,
                                                    float* __restrict__ out) {
  const int t = blockIdx.x;
  const int lane = threadIdx.x & 31;
  float acc[NEXP];
#pragma unroll
  for (int e = 0; e < NEXP; ++e) acc[e] = 0.f;
  for (int k = lane; k < HID_; k += 32) {
    float xv = bf2f(h2[(size_t)t * HID_ + k]);
    const float* wr = w3 + (size_t)k * NEXP;
#pragma unroll
    for (int e = 0; e < NEXP; ++e) acc[e] += xv * wr[e];
  }
#pragma unroll
  for (int off = 16; off; off >>= 1)
#pragma unroll
    for (int e = 0; e < NEXP; ++e) acc[e] += __shfl_xor(acc[e], off);
  if (lane != 0) return;

  float lg[NEXP], p[NEXP];
  float mx = -3.0e38f;
#pragma unroll
  for (int e = 0; e < NEXP; ++e) { lg[e] = acc[e] + b3[e]; mx = fmaxf(mx, lg[e]); }
  float sum = 0.f;
#pragma unroll
  for (int e = 0; e < NEXP; ++e) { p[e] = __expf(lg[e] - mx); sum += p[e]; }
  float inv = 1.0f / sum;
#pragma unroll
  for (int e = 0; e < NEXP; ++e) p[e] *= inv;

  // group scores (4 groups of 2), top-2 groups, top-1 expert among them
  float gs[4];
#pragma unroll
  for (int g = 0; g < 4; ++g) gs[g] = fmaxf(p[2 * g], p[2 * g + 1]);
  int g1 = 0;
#pragma unroll
  for (int g = 1; g < 4; ++g) if (gs[g] > gs[g1]) g1 = g;
  int g2 = -1;
#pragma unroll
  for (int g = 0; g < 4; ++g) if (g != g1 && (g2 < 0 || gs[g] > gs[g2])) g2 = g;
  int best = -1; float bv = -1.f;
  int groups[2] = {g1, g2};
#pragma unroll
  for (int gi = 0; gi < 2; ++gi)
#pragma unroll
    for (int j = 0; j < 2; ++j) {
      int e = 2 * groups[gi] + j;
      if (p[e] > bv) { bv = p[e]; best = e; }
    }
  out[t]          = bv * 1.0f;        // ROUTE_SCALE = 1.0
  out[T_TOK + t]  = (float)best;      // idx
}

// ---------- launcher ----------
extern "C" void kernel_launch(void* const* d_in, const int* in_sizes, int n_in,
                              void* d_out, int out_size, void* d_ws, size_t ws_size,
                              hipStream_t stream) {
  (void)in_sizes; (void)n_in; (void)out_size; (void)ws_size;
  const float* x      = (const float*)d_in[0];   // [T, DIM]
  const float* xp     = (const float*)d_in[1];   // [T, AH]
  const float* anw    = (const float*)d_in[2];   // [AH]
  const float* wq     = (const float*)d_in[3];   // [AH, QKN]
  const float* bq     = (const float*)d_in[4];
  const float* wk     = (const float*)d_in[5];
  const float* bk     = (const float*)d_in[6];
  const float* qnw    = (const float*)d_in[7];   // [HDIM]
  const float* knw    = (const float*)d_in[8];
  const float* w1     = (const float*)d_in[9];   // [DIM, HID]
  const float* b1     = (const float*)d_in[10];
  const float* w2     = (const float*)d_in[11];  // [HID, HID]
  const float* b2     = (const float*)d_in[12];
  const float* w3     = (const float*)d_in[13];  // [HID, E]
  const float* b3     = (const float*)d_in[14];
  float* out = (float*)d_out;                    // [weights(T) | idx(T) | importance(T)]

  char* ws = (char*)d_ws;
  auto alloc = [&](size_t bytes) -> void* {
    void* p = (void*)ws;
    ws += (bytes + 255) & ~(size_t)255;
    return p;
  };
  u16*   wqT   = (u16*)  alloc((size_t)AH_ * QKN * 2);
  u16*   wkT   = (u16*)  alloc((size_t)AH_ * QKN * 2);
  u16*   xn    = (u16*)  alloc((size_t)T_TOK * AH_ * 2);
  float* Qf    = (float*)alloc((size_t)T_TOK * QKN * 4);
  float* Kf    = (float*)alloc((size_t)T_TOK * QKN * 4);
  u16*   Qbf   = (u16*)  alloc((size_t)T_TOK * QKN * 2);
  u16*   Kbf   = (u16*)  alloc((size_t)T_TOK * QKN * 2);
  float* imp_h = (float*)alloc((size_t)NHEAD * T_TOK * 4);
  u16*   xb    = (u16*)  alloc((size_t)T_TOK * DIM_ * 2);
  u16*   w1T   = (u16*)  alloc((size_t)DIM_ * HID_ * 2);
  u16*   w2T   = (u16*)  alloc((size_t)HID_ * HID_ * 2);
  u16*   h1    = (u16*)  alloc((size_t)T_TOK * HID_ * 2);
  u16*   h2    = (u16*)  alloc((size_t)T_TOK * HID_ * 2);

  // weight / activation precision conversion (bf16, weights transposed to N-major)
  transpose_f32_bf16_kernel<<<(AH_ * QKN + 255) / 256, 256, 0, stream>>>(wq, wqT, AH_, QKN);
  transpose_f32_bf16_kernel<<<(AH_ * QKN + 255) / 256, 256, 0, stream>>>(wk, wkT, AH_, QKN);
  transpose_f32_bf16_kernel<<<(DIM_ * HID_ + 255) / 256, 256, 0, stream>>>(w1, w1T, DIM_, HID_);
  transpose_f32_bf16_kernel<<<(HID_ * HID_ + 255) / 256, 256, 0, stream>>>(w2, w2T, HID_, HID_);
  cvt_f32_bf16_kernel<<<(T_TOK * DIM_ + 255) / 256, 256, 0, stream>>>(x, xb, T_TOK * DIM_);

  // attention probe
  rmsnorm_rows_kernel<<<T_TOK, 256, 0, stream>>>(xp, anw, xn, AH_);
  gemm_bf16_kernel<<<dim3(QKN / 64, T_TOK / 128), 256, 0, stream>>>(xn, wqT, bq, Qf, nullptr, QKN, AH_, 0);
  gemm_bf16_kernel<<<dim3(QKN / 64, T_TOK / 128), 256, 0, stream>>>(xn, wkT, bk, Kf, nullptr, QKN, AH_, 0);
  head_rmsnorm_kernel<<<T_TOK * NHEAD, 32, 0, stream>>>(Qf, qnw, Qbf);
  head_rmsnorm_kernel<<<T_TOK * NHEAD, 32, 0, stream>>>(Kf, knw, Kbf);
  zero_f32_kernel<<<(NHEAD * T_TOK + 255) / 256, 256, 0, stream>>>(imp_h, NHEAD * T_TOK);
  scores_imp_kernel<<<dim3(T_TOK / 16, NHEAD), 256, 16 * T_TOK * sizeof(float), stream>>>(Qbf, Kbf, imp_h);
  finalize_imp_kernel<<<(T_TOK + 255) / 256, 256, 0, stream>>>(imp_h, out);

  // router MLP (fused bias + exact GELU epilogues)
  gemm_bf16_kernel<<<dim3(HID_ / 64, T_TOK / 128), 256, 0, stream>>>(xb, w1T, b1, nullptr, h1, HID_, DIM_, 1);
  gemm_bf16_kernel<<<dim3(HID_ / 64, T_TOK / 128), 256, 0, stream>>>(h1, w2T, b2, nullptr, h2, HID_, HID_, 1);
  router_kernel<<<T_TOK, 32, 0, stream>>>(h2, w3, b3, out);
}
